// Attention_29343216566581
// MI455X (gfx1250) — matrix operand order repaired and checked by reading, not compiled, not run on hardware
//
#include <hip/hip_runtime.h>
#include <hip/hip_bf16.h>
#include <cstdint>

typedef __bf16 bf16_t;
typedef __bf16 bf16x4_t  __attribute__((ext_vector_type(4)));
typedef __bf16 bf16x8_t  __attribute__((ext_vector_type(8)));
typedef __bf16 bf16x16_t __attribute__((ext_vector_type(16)));
typedef float  f32x4_t   __attribute__((ext_vector_type(4)));
typedef float  f32x8_t   __attribute__((ext_vector_type(8)));
typedef int    i32x4_t   __attribute__((ext_vector_type(4)));

static constexpr int HID  = 1024;
static constexpr int NH   = 16;
static constexpr int HD   = 64;
static constexpr int SEQ  = 2048;
static constexpr int BATCH= 2;
static constexpr int MTOT = BATCH * SEQ;   // 4096 rows

// ----------------------------------------------------- async LDS copy ----
#if __has_builtin(__builtin_amdgcn_global_load_async_to_lds_b128)
#define HAVE_ASYNC_LDS 1
#else
#define HAVE_ASYNC_LDS 0
#endif

typedef __attribute__((address_space(1))) i32x4_t* glb_v4i_t;
typedef __attribute__((address_space(3))) i32x4_t* lds_v4i_t;

// 16-byte global -> LDS copy. Async (CDNA5 GLOBAL_LOAD_ASYNC_TO_LDS_B128,
// ASYNCcnt-tracked, no VGPR staging) when the builtin exists.
__device__ __forceinline__ void cp16(bf16_t* l, const bf16_t* g) {
#if HAVE_ASYNC_LDS
  __builtin_amdgcn_global_load_async_to_lds_b128(
      (glb_v4i_t)(uintptr_t)g,
      (lds_v4i_t)(uintptr_t)(uint32_t)(uintptr_t)l,
      0, 0);
#else
  *(bf16x8_t*)l = *(const bf16x8_t*)g;
#endif
}

__device__ __forceinline__ void wait_async() {
#if HAVE_ASYNC_LDS
#if __has_builtin(__builtin_amdgcn_s_wait_asynccnt)
  __builtin_amdgcn_s_wait_asynccnt(0);
#else
  asm volatile("s_wait_asynccnt 0x0" ::: "memory");
#endif
#endif
}

// ---------------------------------------------------------------- prep ----
__global__ void k_f32_to_bf16(const float* __restrict__ in,
                              bf16_t* __restrict__ out, int n) {
  int i = (blockIdx.x * blockDim.x + threadIdx.x) * 4;
  if (i < n) {
    f32x4_t v = *(const f32x4_t*)(in + i);
    bf16x4_t o;
    o[0] = (bf16_t)v[0]; o[1] = (bf16_t)v[1];
    o[2] = (bf16_t)v[2]; o[3] = (bf16_t)v[3];
    *(bf16x4_t*)(out + i) = o;
  }
}

// W[k][n] (f32, 1024x1024) -> Wt[n][k] (bf16)
__global__ void k_transpose_w(const float* __restrict__ W,
                              bf16_t* __restrict__ Wt) {
  __shared__ float tile[32][33];
  const int tx = threadIdx.x & 31;
  const int ty = threadIdx.x >> 5;           // 0..7
  const int k0 = blockIdx.y * 32;
  const int n0 = blockIdx.x * 32;
#pragma unroll
  for (int i = 0; i < 32; i += 8)
    tile[ty + i][tx] = W[(size_t)(k0 + ty + i) * HID + n0 + tx];
  __syncthreads();
#pragma unroll
  for (int i = 0; i < 32; i += 8)
    Wt[(size_t)(n0 + ty + i) * HID + k0 + tx] = (bf16_t)tile[tx][ty + i];
}

// ---------------------------------------------------------------- GEMM ----
// C[M=4096][N=1024] = A(bf16, row-major) @ Wt(bf16, N-major)^T + bias
// MODE 0: bf16 out, head layout [b,h,s,d]            (Q, K)
// MODE 1: bf16 out, transposed head layout [b,h,d,s] (V)
// MODE 2: f32 out, row-major [m][n]                  (final projection)
template <int MODE>
__global__ void k_gemm_bf16(const bf16_t* __restrict__ A,
                            const bf16_t* __restrict__ Wt,
                            const float* __restrict__ bias,
                            void* __restrict__ outp) {
  __shared__ bf16_t As[2][128 * 32];
  __shared__ bf16_t Bs[2][128 * 32];

  const int t    = threadIdx.x;
  const int wid  = t >> 5;
  const int lane = t & 31;
  const int r    = lane & 15;
  const int hl   = lane >> 4;
  const int waveM = wid >> 1, waveN = wid & 1;
  const int m_base = blockIdx.y * 128;
  const int n_base = blockIdx.x * 128;
  const int wm0 = waveM * 32, wn0 = waveN * 64;
  const int ldr = t >> 1, ldc = (t & 1) * 16;

  f32x8_t acc[2][4];
#pragma unroll
  for (int fm = 0; fm < 2; ++fm)
#pragma unroll
    for (int fn = 0; fn < 4; ++fn)
      acc[fm][fn] = (f32x8_t){0.f,0.f,0.f,0.f,0.f,0.f,0.f,0.f};

  auto loadTiles = [&](int buf, int k0) {
    const bf16_t* ga = A  + (size_t)(m_base + ldr) * HID + k0 + ldc;
    const bf16_t* gb = Wt + (size_t)(n_base + ldr) * HID + k0 + ldc;
    bf16_t* la = &As[buf][ldr * 32 + ldc];
    bf16_t* lb = &Bs[buf][ldr * 32 + ldc];
    cp16(la,     ga);
    cp16(la + 8, ga + 8);
    cp16(lb,     gb);
    cp16(lb + 8, gb + 8);
  };

  loadTiles(0, 0);
  wait_async();
  __syncthreads();

  for (int kb = 0; kb < HID / 32; ++kb) {
    const int cur = kb & 1;
    if (kb + 1 < HID / 32) loadTiles(cur ^ 1, (kb + 1) * 32);

    bf16x16_t a[2], b[4];
#pragma unroll
    for (int fm = 0; fm < 2; ++fm) {
      const int row = wm0 + fm * 16 + r;
      bf16x8_t lo = *(const bf16x8_t*)&As[cur][row * 32 + hl * 8];
      bf16x8_t hi = *(const bf16x8_t*)&As[cur][row * 32 + hl * 8 + 16];
      a[fm] = __builtin_shufflevector(lo, hi, 0,1,2,3,4,5,6,7,8,9,10,11,12,13,14,15);
    }
#pragma unroll
    for (int fn = 0; fn < 4; ++fn) {
      const int nrow = wn0 + fn * 16 + r;
      bf16x8_t lo = *(const bf16x8_t*)&Bs[cur][nrow * 32 + hl * 8];
      bf16x8_t hi = *(const bf16x8_t*)&Bs[cur][nrow * 32 + hl * 8 + 16];
      b[fn] = __builtin_shufflevector(lo, hi, 0,1,2,3,4,5,6,7,8,9,10,11,12,13,14,15);
    }
#pragma unroll
    for (int fm = 0; fm < 2; ++fm)
#pragma unroll
      for (int fn = 0; fn < 4; ++fn)
        acc[fm][fn] = __builtin_amdgcn_wmma_f32_16x16x32_bf16(
            false, a[fm], false, b[fn], (short)0, acc[fm][fn], false, false);

    wait_async();
    __syncthreads();
  }

#pragma unroll
  for (int fm = 0; fm < 2; ++fm)
#pragma unroll
    for (int fn = 0; fn < 4; ++fn)
#pragma unroll
      for (int j = 0; j < 8; ++j) {
        const int gm = m_base + wm0 + fm * 16 + j + hl * 8;
        const int gn = n_base + wn0 + fn * 16 + r;
        const float v = acc[fm][fn][j] + bias[gn];
        if (MODE == 2) {
          ((float*)outp)[(size_t)gm * HID + gn] = v;
        } else {
          const int bb = gm >> 11, ss = gm & 2047;
          const int hh = gn >> 6,  dd = gn & 63;
          if (MODE == 0)
            ((bf16_t*)outp)[(((size_t)bb * NH + hh) * SEQ + ss) * HD + dd] = (bf16_t)v;
          else
            ((bf16_t*)outp)[(((size_t)bb * NH + hh) * HD + dd) * SEQ + ss] = (bf16_t)v;
        }
      }
}

// ----------------------------------------------------------- attention ----
// One block per (b, h, 64-query tile); 4 waves, each owns 16 queries.
// Computes S^T = K*Q^T and ctx^T = V^T*P^T so K / V^T are natural A-frags,
// Q a natural B-frag, and softmax columns live in a single lane pair.
__global__ void k_attention(const bf16_t* __restrict__ Q,
                            const bf16_t* __restrict__ Kk,
                            const bf16_t* __restrict__ Vt,
                            bf16_t* __restrict__ Ctx) {
  __shared__ bf16_t Ks[2][32 * 64];
  __shared__ bf16_t Vs[2][64 * 32];

  const int t    = threadIdx.x;
  const int wid  = t >> 5;
  const int lane = t & 31;
  const int r    = lane & 15;
  const int hl   = lane >> 4;
  const int bh   = blockIdx.z * NH + blockIdx.y;
  const int q0   = blockIdx.x * 64 + wid * 16;
  const size_t qkBase = (size_t)bh * SEQ * HD;   // [b,h,s,d]
  const size_t vBase  = (size_t)bh * HD * SEQ;   // [b,h,d,s]

  // Q as B-fragments: lane = query column, contiguous d.
  bf16x16_t Qb[2];
#pragma unroll
  for (int c = 0; c < 2; ++c)
    Qb[c] = *(const bf16x16_t*)(Q + qkBase + (size_t)(q0 + r) * HD + c * 32 + hl * 16);

  f32x8_t O[4];
#pragma unroll
  for (int df = 0; df < 4; ++df)
    O[df] = (f32x8_t){0.f,0.f,0.f,0.f,0.f,0.f,0.f,0.f};
  float m_run = -3.0e38f, l_run = 0.f;

  auto loadKV = [&](int buf, int kt) {
    { const int row = t >> 2, co = (t & 3) * 16;
      bf16_t* l = &Ks[buf][row * 64 + co];
      const bf16_t* g = Kk + qkBase + (size_t)(kt + row) * HD + co;
      cp16(l, g); cp16(l + 8, g + 8); }
    { const int row = t >> 1, co = (t & 1) * 16;
      bf16_t* l = &Vs[buf][row * 32 + co];
      const bf16_t* g = Vt + vBase + (size_t)row * SEQ + kt + co;
      cp16(l, g); cp16(l + 8, g + 8); }
  };

  loadKV(0, 0);
  wait_async();
  __syncthreads();

  for (int it = 0; it < SEQ / 32; ++it) {
    const int cur = it & 1;
    if (it + 1 < SEQ / 32) loadKV(cur ^ 1, (it + 1) * 32);

    // ---- load all 4 K A-fragments up front, then 4 score WMMAs ----------
    bf16x16_t ka[4];                      // [kf*2 + c]
#pragma unroll
    for (int kf = 0; kf < 2; ++kf)
#pragma unroll
      for (int c = 0; c < 2; ++c) {
        const int krow = kf * 16 + r;
        bf16x8_t lo = *(const bf16x8_t*)&Ks[cur][krow * 64 + c * 32 + hl * 16];
        bf16x8_t hi = *(const bf16x8_t*)&Ks[cur][krow * 64 + c * 32 + hl * 16 + 8];
        ka[kf * 2 + c] = __builtin_shufflevector(lo, hi, 0,1,2,3,4,5,6,7,8,9,10,11,12,13,14,15);
      }

    f32x8_t st[2];
    st[0] = (f32x8_t){0.f,0.f,0.f,0.f,0.f,0.f,0.f,0.f};
    st[1] = (f32x8_t){0.f,0.f,0.f,0.f,0.f,0.f,0.f,0.f};
#pragma unroll
    for (int kf = 0; kf < 2; ++kf)
#pragma unroll
      for (int c = 0; c < 2; ++c)
        st[kf] = __builtin_amdgcn_wmma_f32_16x16x32_bf16(
            false, ka[kf * 2 + c], false, Qb[c], (short)0, st[kf], false, false);

    // ---- V^T A-fragments loaded early too (hide ds latency under VALU) --
    bf16x16_t va[4];
#pragma unroll
    for (int df = 0; df < 4; ++df) {
      const int drow = df * 16 + r;
      bf16x8_t lo = *(const bf16x8_t*)&Vs[cur][drow * 32 + hl * 16];
      bf16x8_t hi = *(const bf16x8_t*)&Vs[cur][drow * 32 + hl * 16 + 8];
      va[df] = __builtin_shufflevector(lo, hi, 0,1,2,3,4,5,6,7,8,9,10,11,12,13,14,15);
    }

    // ---- online softmax over k (per query column = per lane pair) -------
    float pA[8], pB[8];
    float mx = -3.0e38f;
#pragma unroll
    for (int j = 0; j < 8; ++j) {
      pA[j] = st[0][j] * 0.125f;          // 1/sqrt(64)
      pB[j] = st[1][j] * 0.125f;
      mx = fmaxf(mx, fmaxf(pA[j], pB[j]));
    }
    mx = fmaxf(mx, __shfl_xor(mx, 16, 32));
    const float m_new = fmaxf(m_run, mx);
    const float corr  = __expf(m_run - m_new);
    float sum = 0.f;
#pragma unroll
    for (int j = 0; j < 8; ++j) {
      pA[j] = __expf(pA[j] - m_new);
      pB[j] = __expf(pB[j] - m_new);
      sum += pA[j] + pB[j];
    }
    sum += __shfl_xor(sum, 16, 32);
    l_run = l_run * corr + sum;
    m_run = m_new;
#pragma unroll
    for (int df = 0; df < 4; ++df)
#pragma unroll
      for (int j = 0; j < 8; ++j) O[df][j] *= corr;

    // ---- rebuild P^T as a B-fragment with one half-wave exchange --------
    float ex[8];
#pragma unroll
    for (int j = 0; j < 8; ++j) {
      const float send = (hl == 0) ? pB[j] : pA[j];
      ex[j] = __shfl_xor(send, 16, 32);
    }
    bf16x16_t Pb;
#pragma unroll
    for (int e = 0; e < 16; ++e) {
      const int  i = e & 7;
      const bool lowidx = (e < 8);
      const float v = (hl == 0) ? (lowidx ? pA[i] : ex[i])
                                : (lowidx ? ex[i] : pB[i]);
      Pb[e] = (bf16_t)v;
    }

    // ---- ctx^T += V^T(16d x 32k) * P^T(32k x 16q) -----------------------
#pragma unroll
    for (int df = 0; df < 4; ++df)
      O[df] = __builtin_amdgcn_wmma_f32_16x16x32_bf16(
          false, va[df], false, Pb, (short)0, O[df], false, false);

    wait_async();
    __syncthreads();
  }

  const float rl = 1.0f / l_run;
  const int b_ = blockIdx.z, h_ = blockIdx.y;
#pragma unroll
  for (int df = 0; df < 4; ++df)
#pragma unroll
    for (int j = 0; j < 8; ++j) {
      const int d = df * 16 + j + hl * 8;
      const int q = q0 + r;
      Ctx[((size_t)b_ * SEQ + q) * HID + h_ * HD + d] = (bf16_t)(O[df][j] * rl);
    }
}

// -------------------------------------------------------------- launch ----
extern "C" void kernel_launch(void* const* d_in, const int* in_sizes, int n_in,
                              void* d_out, int out_size, void* d_ws, size_t ws_size,
                              hipStream_t stream) {
  (void)in_sizes; (void)n_in; (void)out_size; (void)ws_size;
  const float* x  = (const float*)d_in[0];
  const float* Wq = (const float*)d_in[1];
  const float* bq = (const float*)d_in[2];
  const float* Wk = (const float*)d_in[3];
  const float* bk = (const float*)d_in[4];
  const float* Wv = (const float*)d_in[5];
  const float* bv = (const float*)d_in[6];
  const float* Wo = (const float*)d_in[7];
  const float* bo = (const float*)d_in[8];
  float* out = (float*)d_out;

  char* ws = (char*)d_ws;
  const size_t MB = (size_t)1 << 20;
  bf16_t* x_bf  = (bf16_t*)(ws + 0);        //  8 MiB: x as bf16
  bf16_t* wq_t  = (bf16_t*)(ws + 8 * MB);   //  2 MiB each: W^T bf16
  bf16_t* wk_t  = (bf16_t*)(ws + 10 * MB);
  bf16_t* wv_t  = (bf16_t*)(ws + 12 * MB);
  bf16_t* wo_t  = (bf16_t*)(ws + 14 * MB);
  bf16_t* q_ws  = (bf16_t*)(ws + 16 * MB);  //  8 MiB [b,h,s,d]
  bf16_t* k_ws  = (bf16_t*)(ws + 24 * MB);  //  8 MiB [b,h,s,d]
  bf16_t* vt_ws = (bf16_t*)(ws + 32 * MB);  //  8 MiB [b,h,d,s]
  bf16_t* c_ws  = (bf16_t*)(ws + 40 * MB);  //  8 MiB [m][hid]

  const int nx = MTOT * HID;                              // 4194304
  k_f32_to_bf16<<<nx / (256 * 4), 256, 0, stream>>>(x, x_bf, nx);

  dim3 tgrid(HID / 32, HID / 32);
  k_transpose_w<<<tgrid, 256, 0, stream>>>(Wq, wq_t);
  k_transpose_w<<<tgrid, 256, 0, stream>>>(Wk, wk_t);
  k_transpose_w<<<tgrid, 256, 0, stream>>>(Wv, wv_t);
  k_transpose_w<<<tgrid, 256, 0, stream>>>(Wo, wo_t);

  dim3 ggrid(HID / 128, MTOT / 128);                      // 8 x 32
  k_gemm_bf16<0><<<ggrid, 256, 0, stream>>>(x_bf, wq_t, bq, (void*)q_ws);
  k_gemm_bf16<0><<<ggrid, 256, 0, stream>>>(x_bf, wk_t, bk, (void*)k_ws);
  k_gemm_bf16<1><<<ggrid, 256, 0, stream>>>(x_bf, wv_t, bv, (void*)vt_ws);

  dim3 agrid(SEQ / 64, NH, BATCH);                        // 32 x 16 x 2
  k_attention<<<agrid, 128, 0, stream>>>(q_ws, k_ws, vt_ws, c_ws);

  k_gemm_bf16<2><<<ggrid, 256, 0, stream>>>(c_ws, wo_t, bo, (void*)out);
}
